// McMotLoss_62002147885409
// MI455X (gfx1250) — compile-verified, matrix operand order checked by reading, not run on hardware
//
#include <hip/hip_runtime.h>
#include <hip/hip_bf16.h>
#include <math.h>
#include <stdint.h>

typedef __attribute__((ext_vector_type(16))) _Float16 v16h;
typedef __attribute__((ext_vector_type(8)))  float    v8f;

#define B_    2
#define C_    5
#define H_    152
#define W_    272
#define K_    128
#define D_    128
#define NID_  300
#define NIDP  304          // NID padded to multiple of 16
#define HW_   (H_*W_)      // 41344
#define N_    (B_*HW_)     // 82688
#define NTILES (N_/16)     // 5168 (exact)
#define NITER  (C_*(NIDP/16))   // 95 staged B tiles
#define EMB_SCALE 8.0616455f   // sqrt(2)*ln(299)

// ---------------- workspace layout ----------------
// acc[0]=pos_sum acc[1]=neg_sum acc[2]=num_pos
// acc[3]=wh_num  acc[4]=off_num acc[5]=mask_sum
// acc[6+c]=ce_sum[c] acc[11+c]=n_valid[c] acc[16+c]=n_elem[c]
// byte 128+ : _Float16 Wh[C_][NIDP][D_]   (f16-packed, zero-padded weights)

__global__ void init_kernel(float* acc) {
    if (threadIdx.x < 32) acc[threadIdx.x] = 0.f;
}

__global__ void wconv_kernel(const float* __restrict__ W, _Float16* __restrict__ Wh) {
    int i = blockIdx.x * blockDim.x + threadIdx.x;
    if (i >= C_ * NIDP * D_) return;
    int d    = i & (D_ - 1);
    int rest = i >> 7;                // / D_
    int nid  = rest % NIDP;
    int c    = rest / NIDP;
    float v = (nid < NID_) ? W[((size_t)c * NID_ + nid) * D_ + d] : 0.f;
    Wh[i] = (_Float16)v;
}

// ---------------- focal loss reduction ----------------
__global__ void focal_kernel(const float* __restrict__ hm, const float* __restrict__ gt,
                             float* acc, int n) {
    int i = blockIdx.x * blockDim.x + threadIdx.x;
    float ps = 0.f, ns = 0.f, np = 0.f;
    if (i < n) {
        float x = hm[i];
        float p = 1.f / (1.f + __expf(-x));
        p = fminf(fmaxf(p, 1e-4f), 1.f - 1e-4f);
        float g = gt[i];
        if (g == 1.f) {
            np = 1.f;
            float q = 1.f - p;
            ps = __logf(p) * q * q;
        } else {
            float w = 1.f - g; w = w * w; w = w * w;   // (1-g)^4
            ns = __logf(1.f - p) * p * p * w;
        }
    }
    #pragma unroll
    for (int m = 1; m < 32; m <<= 1) {
        ps += __shfl_xor(ps, m, 32);
        ns += __shfl_xor(ns, m, 32);
        np += __shfl_xor(np, m, 32);
    }
    __shared__ float sp[8], sn[8], s0[8];
    int w = threadIdx.x >> 5, l = threadIdx.x & 31;
    if (l == 0) { sp[w] = ps; sn[w] = ns; s0[w] = np; }
    __syncthreads();
    if (threadIdx.x == 0) {
        float a = 0.f, b = 0.f, c = 0.f;
        for (int k = 0; k < 8; ++k) { a += sp[k]; b += sn[k]; c += s0[k]; }
        atomicAdd(&acc[0], a); atomicAdd(&acc[1], b); atomicAdd(&acc[2], c);
    }
}

// ---------------- gather L1 (wh + off) ----------------
__global__ void regl1_kernel(const float* __restrict__ wh,  const float* __restrict__ whgt,
                             const float* __restrict__ rg,  const float* __restrict__ rggt,
                             const float* __restrict__ mask, const int* __restrict__ ind,
                             float* acc) {
    int t = threadIdx.x;               // 256 threads, B_*K_ items
    int b = t >> 7, k = t & 127;
    float m  = mask[b * K_ + k];
    int  idx = ind [b * K_ + k];
    float nw = 0.f, no = 0.f;
    #pragma unroll
    for (int ch = 0; ch < 2; ++ch) {
        float pw = wh[((size_t)b * 2 + ch) * HW_ + idx];
        float tw = whgt[((size_t)b * K_ + k) * 2 + ch];
        nw += fabsf(pw * m - tw * m);
        float pr = rg[((size_t)b * 2 + ch) * HW_ + idx];
        float tr = rggt[((size_t)b * K_ + k) * 2 + ch];
        no += fabsf(pr * m - tr * m);
    }
    #pragma unroll
    for (int s = 1; s < 32; s <<= 1) {
        nw += __shfl_xor(nw, s, 32);
        no += __shfl_xor(no, s, 32);
        m  += __shfl_xor(m,  s, 32);
    }
    __shared__ float a0[8], a1[8], a2[8];
    int wv = t >> 5, l = t & 31;
    if (l == 0) { a0[wv] = nw; a1[wv] = no; a2[wv] = m; }
    __syncthreads();
    if (t == 0) {
        float x = 0.f, y = 0.f, z = 0.f;
        for (int i = 0; i < 8; ++i) { x += a0[i]; y += a1[i]; z += a2[i]; }
        acc[3] += x; acc[4] += y; acc[5] += z;   // sole writer after init
    }
}

// ---- fused reid: 4 waves x 16 rows, register-resident A, async double-buffered B ----
__global__ void __launch_bounds__(128)
reid_kernel(const float* __restrict__ idf, const _Float16* __restrict__ Wh,
            const float* __restrict__ clsb, const int* __restrict__ clsmap,
            const int* __restrict__ trids, float* acc) {
    __shared__ __align__(32) _Float16 A[4][16 * 144];    // per-wave A staging (only for setup)
    __shared__ __align__(32) _Float16 Bbuf[2][16 * 128]; // double-buffered shared B tile (4KB)
    __shared__ float lseL[4][16], tlL[4][16];

    const int tid  = threadIdx.x;
    const int w    = tid >> 5;          // wave id: rows w*16..w*16+15 of this block
    const int lane = tid & 31;
    const int g    = lane >> 4;         // half-group: fragment rows g*8..g*8+7
    const int lm   = lane & 15;
    const int rowbase = blockIdx.x * 64 + w * 16;

    // ---- load + L2-normalize this wave's 16x128 A tile into LDS ----
    {
        int n  = rowbase + lm;
        int b  = n / HW_;
        int hw = n - b * HW_;
        const float* src = idf + (size_t)b * D_ * HW_ + hw;
        int dbase = g * 64;
        float ss = 0.f;
        for (int d = 0; d < 64; ++d) {
            float v = src[(size_t)(dbase + d) * HW_];
            ss += v * v;
        }
        ss += __shfl_xor(ss, 16, 32);
        float scl = EMB_SCALE / fmaxf(sqrtf(ss), 1e-12f);
        for (int d = 0; d < 64; ++d) {
            float v = src[(size_t)(dbase + d) * HW_];
            A[w][lm * 144 + dbase + d] = (_Float16)(v * scl);
        }
    }

    int n0  = rowbase + lm;
    int b0  = n0 / HW_;
    int hw0 = n0 - b0 * HW_;
    int cm  = clsmap[b0 * HW_ + hw0];   // class map for row (lane&15); lanes 16-31 mirror 0-15

    const uint32_t bb0 = (uint32_t)(uintptr_t)&Bbuf[0][0];
    const uint32_t bb1 = (uint32_t)(uintptr_t)&Bbuf[1][0];

    // stage tile 0 (c=0,j=0): 128 threads x 32B = 4KB, two async b128 ops per thread
    {
        uint64_t gsrc = (uint64_t)(uintptr_t)Wh + (uint64_t)tid * 32;
        uint32_t l0 = bb0 + (uint32_t)tid * 32;
        asm volatile("global_load_async_to_lds_b128 %0, %1, off"
                     :: "v"(l0), "v"(gsrc) : "memory");
        asm volatile("global_load_async_to_lds_b128 %0, %1, off"
                     :: "v"(l0 + 16u), "v"(gsrc + 16u) : "memory");
    }

    // ---- hoist A fragments into registers once (survive all later barriers) ----
    __syncthreads();                    // A tile visible to all lanes of the wave
    v16h afrag[4];
    #pragma unroll
    for (int s = 0; s < 4; ++s) {
        int koff = 32 * s + 16 * g;
        afrag[s] = *reinterpret_cast<const v16h*>(&A[w][lm * 144 + koff]);
    }

    int buf = 0, it = 0;
    for (int c = 0; c < C_; ++c) {
        int t  = trids[((size_t)b0 * C_ + c) * HW_ + hw0];   // target for row (lane&15)
        int tc = t < 0 ? 0 : (t > NID_ - 1 ? NID_ - 1 : t);
        int myT[8];
        float rmax[8], rsum[8], tl[8];
        #pragma unroll
        for (int r = 0; r < 8; ++r) {
            myT[r]  = __shfl(tc, g * 8 + r, 32);   // clamped target of fragment row r
            rmax[r] = -1e30f; rsum[r] = 0.f; tl[r] = 0.f;
        }

        for (int j = 0; j < NIDP / 16; ++j) {
            // software pipeline: prefetch next tile into other buffer, counted wait
            int nxt = it + 1;
            if (nxt < NITER) {
                int cc = nxt / (NIDP / 16), jj = nxt - cc * (NIDP / 16);
                uint64_t gsrc = (uint64_t)(uintptr_t)(Wh + ((size_t)cc * NIDP + jj * 16) * D_)
                              + (uint64_t)tid * 32;
                uint32_t l0 = (buf ? bb0 : bb1) + (uint32_t)tid * 32;
                asm volatile("global_load_async_to_lds_b128 %0, %1, off"
                             :: "v"(l0), "v"(gsrc) : "memory");
                asm volatile("global_load_async_to_lds_b128 %0, %1, off"
                             :: "v"(l0 + 16u), "v"(gsrc + 16u) : "memory");
                asm volatile("s_wait_asynccnt 0x2" ::: "memory"); // prev tile's 2 ops done
            } else {
                asm volatile("s_wait_asynccnt 0x0" ::: "memory");
            }
            __syncthreads();

            const _Float16* Bb = &Bbuf[buf][0];
            v8f accv = {};
            #pragma unroll
            for (int s = 0; s < 4; ++s) {            // K = 128 = 4 x 32
                int koff = 32 * s + 16 * g;
                v16h bv = *reinterpret_cast<const v16h*>(&Bb[lm * 128 + koff]);
                accv = __builtin_amdgcn_wmma_f32_16x16x32_f16(
                    false, afrag[s], false, bv, (short)0, accv, false, false);
            }
            int  col = j * 16 + lm;
            bool cv  = col < NID_;
            float bias = cv ? clsb[c * NID_ + col] : 0.f;
            #pragma unroll
            for (int r = 0; r < 8; ++r) {
                float logit = cv ? (accv[r] + bias) : -1e30f;
                float nm = fmaxf(rmax[r], logit);
                rsum[r] = rsum[r] * __expf(rmax[r] - nm) + __expf(logit - nm);
                rmax[r] = nm;
                if (col == myT[r]) tl[r] += accv[r] + bias;
            }
            buf ^= 1; ++it;
        }

        // reduce (max, sumexp, target-logit) across the 16 lanes of each half-group
        #pragma unroll
        for (int m = 1; m < 16; m <<= 1) {
            #pragma unroll
            for (int r = 0; r < 8; ++r) {
                float om = __shfl_xor(rmax[r], m, 32);
                float os = __shfl_xor(rsum[r], m, 32);
                float nm = fmaxf(rmax[r], om);
                rsum[r] = rsum[r] * __expf(rmax[r] - nm) + os * __expf(om - nm);
                rmax[r] = nm;
                tl[r]  += __shfl_xor(tl[r], m, 32);
            }
        }
        if (lm == 0) {
            #pragma unroll
            for (int r = 0; r < 8; ++r) {
                lseL[w][g * 8 + r] = rmax[r] + __logf(rsum[r]);
                tlL [w][g * 8 + r] = tl[r];
            }
        }
        __syncthreads();

        float ceV = 0.f, vV = 0.f, eV = 0.f;
        if (g == 0) {
            bool isC   = (cm == c);
            bool valid = isC && (t != -1);
            eV  = isC   ? 1.f : 0.f;
            vV  = valid ? 1.f : 0.f;
            ceV = valid ? (lseL[w][lm] - tlL[w][lm]) : 0.f;
        }
        #pragma unroll
        for (int m = 1; m < 32; m <<= 1) {
            ceV += __shfl_xor(ceV, m, 32);
            vV  += __shfl_xor(vV,  m, 32);
            eV  += __shfl_xor(eV,  m, 32);
        }
        if (lane == 0) {
            atomicAdd(&acc[6  + c], ceV);
            atomicAdd(&acc[11 + c], vV);
            atomicAdd(&acc[16 + c], eV);
        }
        // next class's first j-iteration barrier orders lseL reads vs next writes
    }
}

// ---------------- finalize ----------------
__global__ void fin_kernel(const float* acc, const float* sdet, const float* sid, float* out) {
    float pos = acc[0], neg = acc[1], np = acc[2];
    float hm  = (np > 0.f) ? -(pos + neg) / fmaxf(np, 1.f) : -neg;
    float den = acc[5] * 2.f + 1e-4f;
    float whl = acc[3] / den;
    float offl = acc[4] / den;
    float reid = 0.f;
    for (int c = 0; c < C_; ++c) {
        float cem = acc[6 + c] / fmaxf(acc[11 + c], 1.f);
        float ne  = acc[16 + c];
        reid += (ne > 0.f) ? cem / fmaxf(ne, 1.f) : 0.f;
    }
    float det = 1.0f * hm + 0.1f * whl + 1.0f * offl;
    float sd = sdet[0], si = sid[0];
    float loss = 0.5f * (__expf(-sd) * det + __expf(-si) * reid + sd + si);
    out[0] = loss; out[1] = hm; out[2] = whl; out[3] = offl; out[4] = reid;
}

extern "C" void kernel_launch(void* const* d_in, const int* in_sizes, int n_in,
                              void* d_out, int out_size, void* d_ws, size_t ws_size,
                              hipStream_t stream) {
    const float* hm        = (const float*)d_in[0];
    const float* hm_gt     = (const float*)d_in[1];
    const float* wh        = (const float*)d_in[2];
    const float* wh_gt     = (const float*)d_in[3];
    const float* reg       = (const float*)d_in[4];
    const float* reg_gt    = (const float*)d_in[5];
    const float* id_feat   = (const float*)d_in[6];
    const float* cls_W     = (const float*)d_in[7];
    const float* cls_b     = (const float*)d_in[8];
    const float* s_det     = (const float*)d_in[9];
    const float* s_id      = (const float*)d_in[10];
    const float* reg_mask  = (const float*)d_in[11];
    const int*   ind       = (const int*)d_in[12];
    const int*   cls_id_map = (const int*)d_in[13];
    const int*   cls_tr_ids = (const int*)d_in[14];

    float*    acc = (float*)d_ws;
    _Float16* Wh  = (_Float16*)((char*)d_ws + 128);

    init_kernel<<<1, 32, 0, stream>>>(acc);

    int nw = C_ * NIDP * D_;
    wconv_kernel<<<(nw + 255) / 256, 256, 0, stream>>>(cls_W, Wh);

    int nh = B_ * C_ * H_ * W_;
    focal_kernel<<<(nh + 255) / 256, 256, 0, stream>>>(hm, hm_gt, acc, nh);

    regl1_kernel<<<1, 256, 0, stream>>>(wh, wh_gt, reg, reg_gt, reg_mask, ind, acc);

    reid_kernel<<<NTILES / 4, 128, 0, stream>>>(id_feat, Wh, cls_b, cls_id_map, cls_tr_ids, acc);

    fin_kernel<<<1, 1, 0, stream>>>(acc, s_det, s_id, (float*)d_out);
}
